// TreeLSTMSTS_65077344469014
// MI455X (gfx1250) — compile-verified, matrix operand order checked by reading, not run on hardware
//
#include <hip/hip_runtime.h>
#include <hip/hip_bf16.h>

typedef unsigned short u16;
typedef __attribute__((ext_vector_type(16))) __bf16 v16bf;
typedef __attribute__((ext_vector_type(8)))  float  v8f;
typedef __attribute__((ext_vector_type(8)))  unsigned int v8u;

#define NPAIR 16      // (batch,tree) pairs per workgroup
#define KTOT  1536    // concatenated K = D + H
#define LDA   1544    // padded ushort row stride for A tiles (breaks LDS bank conflicts)
#define LDF   772     // padded float row stride for fc tile
#define NNODE 256
#define HDIM  768

__device__ __forceinline__ u16 f2bf(float x) {
    unsigned int u = __builtin_bit_cast(unsigned int, x);
    unsigned int r = u + 0x7FFFu + ((u >> 16) & 1u);   // round-to-nearest-even
    return (u16)(r >> 16);
}
__device__ __forceinline__ float sigm(float x) { return 1.f / (1.f + __expf(-x)); }

// Async DMA: 16 bytes per lane, global -> LDS, tracked by ASYNCcnt (ISA §15.18.3 op 98).
// lds_off is the raw LDS byte address (low 32 bits of a generic shared pointer).
__device__ __forceinline__ void async_g2l_b128(unsigned lds_off, const void* gptr) {
    asm volatile("global_load_async_to_lds_b128 %0, %1, off"
                 :: "v"(lds_off), "v"(gptr)
                 : "memory");
}
__device__ __forceinline__ void wait_async() {
    asm volatile("s_wait_asynccnt 0" ::: "memory");
}
__device__ __forceinline__ unsigned lds_addr_of(const void* p) {
    return (unsigned)(unsigned long long)p;   // generic LDS ptr: low 32 bits = LDS byte offset
}

// Load one 16x32 bf16 WMMA operand fragment (A or B) for this lane.
// Layout per ISA 7.12.2: lanes 0-15 hold K[kb+0..7] & K[kb+16..23] of row (lane&15),
// lanes 16-31 hold K[kb+8..15] & K[kb+24..31]. Caller pre-adds row*ld + kb + kh.
__device__ __forceinline__ v16bf load_frag(const u16* p) {
    uint4 lo = *(const uint4*)(p);
    uint4 hi = *(const uint4*)(p + 16);
    v8u u;
    u[0]=lo.x; u[1]=lo.y; u[2]=lo.z; u[3]=lo.w;
    u[4]=hi.x; u[5]=hi.y; u[6]=hi.z; u[7]=hi.w;
    return __builtin_bit_cast(v16bf, u);
}

// ---------------------------------------------------------------- prep kernels
__global__ void zero_kernel(float* __restrict__ p, long n) {
    long i = (long)blockIdx.x * blockDim.x + threadIdx.x;
    long stride = (long)gridDim.x * blockDim.x;
    for (; i < n; i += stride) p[i] = 0.f;
}

__global__ void cvt_kernel(const float* __restrict__ a, const float* __restrict__ b,
                           u16* __restrict__ out, long half) {
    long i = (long)blockIdx.x * blockDim.x + threadIdx.x;
    long stride = (long)gridDim.x * blockDim.x;
    long n = 2 * half;
    for (; i < n; i += stride) {
        float v = (i < half) ? a[i] : b[i - half];
        out[i] = f2bf(v);
    }
}

// Pack [W_iou | U_iou] -> wciou (2304 x 1536 bf16) and [W_f | U_f] -> wcf (768 x 1536 bf16)
__global__ void packw_kernel(const float* __restrict__ Wiou, const float* __restrict__ Uiou,
                             const float* __restrict__ Wf,   const float* __restrict__ Uf,
                             u16* __restrict__ wciou, u16* __restrict__ wcf) {
    long n1 = 2304L * KTOT;
    long ntot = n1 + 768L * KTOT;
    long i = (long)blockIdx.x * blockDim.x + threadIdx.x;
    long stride = (long)gridDim.x * blockDim.x;
    for (; i < ntot; i += stride) {
        if (i < n1) {
            long n = i / KTOT, k = i % KTOT;
            float v = (k < HDIM) ? Wiou[n * HDIM + k] : Uiou[n * HDIM + (k - HDIM)];
            wciou[i] = f2bf(v);
        } else {
            long r = i - n1, n = r / KTOT, k = r % KTOT;
            float v = (k < HDIM) ? Wf[n * HDIM + k] : Uf[n * HDIM + (k - HDIM)];
            wcf[r] = f2bf(v);
        }
    }
}

// ---------------------------------------------------------------- recurrent scan
// 8 workgroups x 512 threads (16 waves). WG g owns pairs q = g*16..g*16+15
// (q<64 -> tree A, else tree B). Per step i = 255..0:
//   GEMM1: [x[i] | h_sum[i]] (16x1536) @ wciou^T -> gates (16x2304)
//   gates -> c, hi (register-resident, WMMA C layout)
//   GEMM2: [x[parent] | hi] (16x1536) @ wcf^T -> f
//   scatter h_sum[parent] += hi ; fc_sum[parent] += f*c
__global__ void __launch_bounds__(512)
scan_kernel(const u16* __restrict__ xbf,
            const u16* __restrict__ wciou,
            const u16* __restrict__ wcf,
            const float* __restrict__ b_iou,
            const float* __restrict__ b_f,
            const int* __restrict__ parent_a,
            const int* __restrict__ parent_b,
            float* __restrict__ hsum,
            float* __restrict__ fcsum,
            float* __restrict__ rroot) {
    extern __shared__ unsigned char smem[];
    u16*   sA1  = (u16*)smem;                 // 16 x LDA  bf16   ([x[i] | h_sum[i]])
    u16*   sA2  = sA1 + 16 * LDA;             // 16 x LDA  bf16   ([x[parent] | hi])
    float* sfc  = (float*)(sA2 + 16 * LDA);   // 16 x LDF  f32    (fc_sum[i])
    int*   spar = (int*)(sfc + 16 * LDF);     // 16               (parent[i] per pair)

    const int tid  = threadIdx.x;
    const int w    = tid >> 5;
    const int lane = tid & 31;
    const int wg   = blockIdx.x;
    const int* parent = (wg >= 4) ? parent_b : parent_a;

    const int n16 = lane & 15;          // N column / A row within tile
    const int kh  = (lane >> 4) * 8;    // K sub-offset for this half-wave
    const int mlo = (lane >> 4) * 8;    // C-layout: m = j + mlo
    const int hb  = w * 48 + n16;       // hidden-unit base column for this wave/lane

    // Loop-invariant per-lane biases (one per owned N-tile)
    float bi[3], bo[3], bu[3], bfv[3];
#pragma unroll
    for (int tt = 0; tt < 3; ++tt) {
        int h = hb + tt * 16;
        bi[tt]  = b_iou[h];
        bo[tt]  = b_iou[HDIM + h];
        bu[tt]  = b_iou[2 * HDIM + h];
        bfv[tt] = b_f[h];
    }

    for (int i = NNODE - 1; i >= 0; --i) {
        // ---- phase 1a: async-DMA x[i] rows into A1 low half (96 b128 chunks/row)
        for (int idx = tid; idx < NPAIR * 96; idx += 512) {
            int m = idx / 96, c = idx % 96;
            int q = wg * NPAIR + m;
            async_g2l_b128(lds_addr_of(sA1 + m * LDA) + c * 16,
                           xbf + ((long)q * NNODE + i) * HDIM + c * 8);
        }
        // ---- phase 1b: stage bf16(h_sum[i]) into A1 high half + fc row (needs cvt)
        for (int idx = tid; idx < NPAIR * HDIM; idx += 512) {
            int m = idx / HDIM, k = idx % HDIM;
            int q = wg * NPAIR + m;
            long base = ((long)q * NNODE + i) * HDIM + k;
            sA1[m * LDA + HDIM + k] = f2bf(hsum[base]);
            sfc[m * LDF + k] = fcsum[base];
        }
        if (tid < NPAIR) {
            int q = wg * NPAIR + tid;
            spar[tid] = parent[(q & 63) * NNODE + i];
        }
        wait_async();
        __syncthreads();

        // ---- async-DMA x[parent] rows into A2 low half; completion overlapped
        //      with GEMM1 + gate math, consumed only after the next barrier.
        for (int idx = tid; idx < NPAIR * 96; idx += 512) {
            int m = idx / 96, c = idx % 96;
            int q = wg * NPAIR + m;
            async_g2l_b128(lds_addr_of(sA2 + m * LDA) + c * 16,
                           xbf + ((long)q * NNODE + spar[m]) * HDIM + c * 8);
        }

        // ---- GEMM1: 9 N-tiles (3 gates x 3 column tiles) per wave, K = 1536
        v8f acc[9] = {};
        for (int kb = 0; kb < KTOT; kb += 32) {
            v16bf afrag = load_frag(sA1 + n16 * LDA + kb + kh);
            __builtin_prefetch(wciou + (long)(hb) * KTOT + kb + 64, 0, 1);
#pragma unroll
            for (int g3 = 0; g3 < 3; ++g3) {
#pragma unroll
                for (int tt = 0; tt < 3; ++tt) {
                    int col = g3 * HDIM + w * 48 + tt * 16 + n16;
                    v16bf bfrag = load_frag(wciou + (long)col * KTOT + kb + kh);
                    acc[g3 * 3 + tt] = __builtin_amdgcn_wmma_f32_16x16x32_bf16(
                        false, afrag, false, bfrag, (short)0, acc[g3 * 3 + tt], false, false);
                }
            }
        }

        // ---- gates -> c, hi  (i/o/u for one hidden unit share lane+VGPR slot)
        float creg[3][8], hreg[3][8];
#pragma unroll
        for (int tt = 0; tt < 3; ++tt) {
            int h = hb + tt * 16;
#pragma unroll
            for (int j = 0; j < 8; ++j) {
                int m = j + mlo;
                float ig = acc[0 * 3 + tt][j] + bi[tt];
                float og = acc[1 * 3 + tt][j] + bo[tt];
                float ug = acc[2 * 3 + tt][j] + bu[tt];
                float cv = sigm(ig) * tanhf(ug) + sfc[m * LDF + h];
                float hv = sigm(og) * tanhf(cv);
                creg[tt][j] = cv;
                hreg[tt][j] = hv;
                sA2[m * LDA + HDIM + h] = f2bf(hv);
                if (i == 0) rroot[(long)(wg * NPAIR + m) * HDIM + h] = hv;  // root repr
            }
        }
        wait_async();       // A2 low half (x[parent]) landed in LDS
        __syncthreads();

        // ---- GEMM2: f_pre = [x[p] | hi] @ wcf^T, 3 N-tiles per wave
        v8f acc2[3] = {};
        for (int kb = 0; kb < KTOT; kb += 32) {
            v16bf afrag = load_frag(sA2 + n16 * LDA + kb + kh);
#pragma unroll
            for (int tt = 0; tt < 3; ++tt) {
                int col = w * 48 + tt * 16 + n16;
                v16bf bfrag = load_frag(wcf + (long)col * KTOT + kb + kh);
                acc2[tt] = __builtin_amdgcn_wmma_f32_16x16x32_bf16(
                    false, afrag, false, bfrag, (short)0, acc2[tt], false, false);
            }
        }

        // ---- f = sigmoid(f_pre + b_f); scatter to parent rows (disjoint per lane)
#pragma unroll
        for (int tt = 0; tt < 3; ++tt) {
            int h = hb + tt * 16;
#pragma unroll
            for (int j = 0; j < 8; ++j) {
                int m = j + mlo;
                int q = wg * NPAIR + m;
                int p = spar[m];
                float f = sigm(acc2[tt][j] + bfv[tt]);
                long base = ((long)q * NNODE + p) * HDIM + h;
                hsum[base]  += hreg[tt][j];
                fcsum[base] += f * creg[tt][j];
            }
        }
        __threadfence();
        __syncthreads();
    }
}

// ---------------------------------------------------------------- similarity head
__global__ void head_kernel(const float* __restrict__ rroot,
                            const float* __restrict__ Wmul,  const float* __restrict__ bmul,
                            const float* __restrict__ Wdiff, const float* __restrict__ bdiff,
                            const float* __restrict__ Wpred, const float* __restrict__ bpred,
                            float* __restrict__ out) {
    __shared__ float hs[1024];
    __shared__ float red[128];
    __shared__ float lg[8];
    const int b = blockIdx.x, t = threadIdx.x;
    const float* r1 = rroot + (long)b * HDIM;
    const float* r2 = rroot + (long)(64 + b) * HDIM;

    for (int o = t; o < 1024; o += 128) {
        bool mul = (o < 512);
        int row = mul ? o : (o - 512);
        const float* W = mul ? Wmul : Wdiff;
        float s = mul ? bmul[row] : bdiff[row];
        for (int k = 0; k < HDIM; ++k) {
            float a1 = r1[k], a2 = r2[k];
            float v = mul ? (a1 * a2) : fabsf(a1 - a2);
            s += W[row * HDIM + k] * v;
        }
        hs[o] = sigm(s);
    }
    __syncthreads();

    for (int r = 0; r < 6; ++r) {
        float p = 0.f;
        for (int k = t; k < 1024; k += 128) p += hs[k] * Wpred[r * 1024 + k];
        red[t] = p;
        __syncthreads();
        for (int s = 64; s > 0; s >>= 1) {
            if (t < s) red[t] += red[t + s];
            __syncthreads();
        }
        if (t == 0) lg[r] = red[0] + bpred[r];
        __syncthreads();
    }
    if (t == 0) {
        float mx = lg[0];
        for (int r = 1; r < 6; ++r) mx = fmaxf(mx, lg[r]);
        float se = 0.f;
        for (int r = 0; r < 6; ++r) se += __expf(lg[r] - mx);
        float ls = logf(se);
        for (int r = 0; r < 6; ++r) out[b * 6 + r] = lg[r] - mx - ls;
    }
}

// ---------------------------------------------------------------- launcher
extern "C" void kernel_launch(void* const* d_in, const int* in_sizes, int n_in,
                              void* d_out, int out_size, void* d_ws, size_t ws_size,
                              hipStream_t stream) {
    const float* emb1 = (const float*)d_in[0];
    const float* emb2 = (const float*)d_in[1];
    const int* parent_a = (const int*)d_in[2];
    const int* parent_b = (const int*)d_in[3];
    const float* W_iou = (const float*)d_in[4];
    const float* U_iou = (const float*)d_in[5];
    const float* b_iou = (const float*)d_in[6];
    const float* W_f   = (const float*)d_in[7];
    const float* U_f   = (const float*)d_in[8];
    const float* b_f   = (const float*)d_in[9];
    const float* W_mul = (const float*)d_in[10];
    const float* b_mul = (const float*)d_in[11];
    const float* W_diff= (const float*)d_in[12];
    const float* b_diff= (const float*)d_in[13];
    const float* W_pred= (const float*)d_in[14];
    const float* b_pred= (const float*)d_in[15];
    float* out = (float*)d_out;

    // workspace layout (bytes, all 256-aligned)
    const long N_X   = 2L * 64 * NNODE * HDIM;          // 25,165,824 elements
    char* ws = (char*)d_ws;
    u16*   xbf   = (u16*)(ws);                          // 50,331,648 B
    u16*   wciou = (u16*)(ws + 50331648L);              //  7,077,888 B
    u16*   wcf   = (u16*)(ws + 57409536L);              //  2,359,296 B
    float* hsum  = (float*)(ws + 59768832L);            // 100,663,296 B
    float* fcsum = (float*)(ws + 160432128L);           // 100,663,296 B
    float* rroot = (float*)(ws + 261095424L);           //     393,216 B

    // state must be re-zeroed every call (harness replays without resetting ws)
    zero_kernel<<<2048, 256, 0, stream>>>(hsum, 2L * 128 * NNODE * HDIM);
    cvt_kernel<<<2048, 256, 0, stream>>>(emb1, emb2, xbf, N_X / 2);
    packw_kernel<<<1024, 256, 0, stream>>>(W_iou, U_iou, W_f, U_f, wciou, wcf);

    const size_t smem = (size_t)(2 * 16 * LDA) * sizeof(u16)
                      + (size_t)(16 * LDF) * sizeof(float) + 64;   // 148,288 B (<320KB/WGP)
    scan_kernel<<<8, 512, smem, stream>>>(xbf, wciou, wcf, b_iou, b_f,
                                          parent_a, parent_b, hsum, fcsum, rroot);

    head_kernel<<<64, 128, 0, stream>>>(rroot, W_mul, b_mul, W_diff, b_diff,
                                        W_pred, b_pred, out);
}
